// MultiHeadCrossGraph_37606733644542
// MI455X (gfx1250) — matrix operand order repaired and checked by reading, not compiled, not run on hardware
//
#include <hip/hip_runtime.h>
#include <hip/hip_bf16.h>
#include <math.h>
#include <stdint.h>

// Problem dims
constexpr int nB = 8, nN = 128, nIN = 256, nH = 4, nD = 64, nM = 129;
constexpr int nMP = 160;          // M padded to multiple of 32 (WMMA K)
constexpr int nHD = nH * nD;      // 256
constexpr int SLAB = nB * nN * nHD;  // 262144 floats per projection slab

typedef __attribute__((ext_vector_type(16))) _Float16 v16h;
typedef __attribute__((ext_vector_type(8)))  float    v8f;
typedef _Float16 f16;

// ---------------- WMMA fragment helpers (CDNA5 16x16x32 f16 layouts) -------
// A 16x32: lane L holds row (L&15); k = kb + {0..15}, kb = (L>=16)?16:0
static __device__ __forceinline__ v16h a_frag_f32(const float* rowp, int kk, int lane) {
  const int kb = (lane & 16) ? 16 : 0;
  const float* p = rowp + kk + kb;
  v16h a;
#pragma unroll
  for (int i = 0; i < 8; ++i) a[i] = (f16)p[i];
#pragma unroll
  for (int i = 0; i < 8; ++i) a[8 + i] = (f16)p[8 + i];
  return a;
}
static __device__ __forceinline__ v16h a_frag_f16(const f16* rowp, int kk, int lane) {
  const int kb = (lane & 16) ? 16 : 0;
  const f16* p = rowp + kk + kb;
  v16h a;
#pragma unroll
  for (int i = 0; i < 8; ++i) a[i] = p[i];
#pragma unroll
  for (int i = 0; i < 8; ++i) a[8 + i] = p[8 + i];
  return a;
}
// B 32x16: lane L holds col (L&15); k = kb + i, kb = (L>=16)?16:0
static __device__ __forceinline__ v16h b_frag_f32(const float* colp, int kk, int lane) {
  const int kb = (lane & 16) ? 16 : 0;
  const float* p = colp + kk + kb;
  v16h b;
#pragma unroll
  for (int i = 0; i < 16; ++i) b[i] = (f16)p[i];
  return b;
}
static __device__ __forceinline__ v16h b_frag_f16(const f16* colp, int kk, int lane) {
  const int kb = (lane & 16) ? 16 : 0;
  const f16* p = colp + kk + kb;
  v16h b;
#pragma unroll
  for (int i = 0; i < 16; ++i) b[i] = p[i];
  return b;
}
// B from LDS-resident aa_w (row-major [129][129] f32): B[k=m][o] = AAW[o*129+m],
// zero outside the 129x129 valid region (cols/k padded to 160).
static __device__ __forceinline__ v16h b_frag_aa_lds(const float* AAW, int og, int kk, int lane) {
  const int k0 = kk + ((lane & 16) ? 16 : 0);
  v16h b;
#pragma unroll
  for (int i = 0; i < 16; ++i) {
    const int m = k0 + i;
    b[i] = (og < nM && m < nM) ? (f16)AAW[og * nM + m] : (f16)0.0f;
  }
  return b;
}
// B row-major with leading dim (k strided): B[k][j] = base[k*ld + j]
static __device__ __forceinline__ v16h b_frag_f16_strided(const f16* base, int ld, int col, int kk, int lane) {
  const int k0 = kk + ((lane & 16) ? 16 : 0);
  const f16* p = base + (size_t)k0 * ld + col;
  v16h b;
#pragma unroll
  for (int i = 0; i < 16; ++i) b[i] = p[(size_t)i * ld];
  return b;
}
static __device__ __forceinline__ v8f wmma_f16(v16h a, v16h b, v8f c) {
  return __builtin_amdgcn_wmma_f32_16x16x32_f16(false, a, false, b, (short)0, c, false, false);
}

// gfx1250 async global->LDS DMA (ASYNCcnt-tracked; bypasses VGPRs)
static __device__ __forceinline__ void async_g2lds_b128(uint32_t lds_off, const void* gptr) {
  asm volatile("global_load_async_to_lds_b128 %0, %1, off" :: "v"(lds_off), "v"(gptr) : "memory");
}
static __device__ __forceinline__ void async_g2lds_b32(uint32_t lds_off, const void* gptr) {
  asm volatile("global_load_async_to_lds_b32 %0, %1, off" :: "v"(lds_off), "v"(gptr) : "memory");
}
static __device__ __forceinline__ void wait_async0() {
  asm volatile("s_wait_asynccnt 0x0" ::: "memory");
}

// ---------------- Kernel 1: six projection GEMMs  P = X * W^T + bias -------
// slabs: 0 PK1(m1,k) 1 PK2(m2,k) 2 PQ1(m1,q) 3 PQ2(m2,q) 4 PV1(m1,v) 5 PV2(m2,v)
__global__ __launch_bounds__(256) void proj_gemm_kernel(
    const float* __restrict__ m1, const float* __restrict__ m2,
    const float* __restrict__ kw, const float* __restrict__ kb_,
    const float* __restrict__ qw, const float* __restrict__ qb_,
    const float* __restrict__ vw, const float* __restrict__ vb_,
    float* __restrict__ P) {
  const int lane = threadIdx.x & 31, wv = threadIdx.x >> 5;
  const int rt = blockIdx.x;              // 0..63 row tiles over 1024 rows
  const int ct = blockIdx.y * 8 + wv;     // 0..15 col tiles over 256 cols
  const int which = blockIdx.z;           // 0..5
  const float* X = (which & 1) ? m2 : m1;
  const int sel = which >> 1;
  const float* W    = (sel == 0) ? kw  : (sel == 1) ? qw  : vw;
  const float* bias = (sel == 0) ? kb_ : (sel == 1) ? qb_ : vb_;
  float* out = P + (size_t)which * SLAB;

  const int row = rt * 16 + (lane & 15);
  const int col = ct * 16 + (lane & 15);
  const float* rowp = X + (size_t)row * nIN;
  const float* colp = W + (size_t)col * nIN;

  v8f acc = {};
#pragma unroll
  for (int kk = 0; kk < nIN; kk += 32)
    acc = wmma_f16(a_frag_f32(rowp, kk, lane), b_frag_f32(colp, kk, lane), acc);

  const int rbase = rt * 16 + ((lane & 16) ? 8 : 0);
#pragma unroll
  for (int r = 0; r < 8; ++r)
    out[(size_t)(rbase + r) * nHD + col] = acc[r] + bias[col];
}

// ---------------- Kernel 2: joint LN stats for K and Q (per batch) ---------
__global__ __launch_bounds__(256) void stats_kq_kernel(const float* __restrict__ P,
                                                       float* __restrict__ statsKQ) {
  const int b = blockIdx.x, which = blockIdx.y;  // 0=K, 1=Q
  const float* P1 = P + (size_t)(which * 2 + 0) * SLAB + (size_t)b * nN * nHD;
  const float* P2 = P + (size_t)(which * 2 + 1) * SLAB + (size_t)b * nN * nHD;
  float s = 0.f, ss = 0.f;
  for (int i = threadIdx.x; i < nN * nHD; i += 256) {
    const float v1 = P1[i], v2 = P2[i];
    s  += v1 + 128.0f * v2;          // m2 rows appear N=128 times (once per f)
    ss += v1 * v1 + 128.0f * v2 * v2;
  }
  __shared__ float ra[256], rb[256];
  ra[threadIdx.x] = s; rb[threadIdx.x] = ss;
  __syncthreads();
  for (int st = 128; st > 0; st >>= 1) {
    if (threadIdx.x < st) { ra[threadIdx.x] += ra[threadIdx.x + st]; rb[threadIdx.x] += rb[threadIdx.x + st]; }
    __syncthreads();
  }
  if (threadIdx.x == 0) {
    const float cnt = (float)nN * (float)nM * (float)nHD;  // 128*129*256
    const float mu = ra[0] / cnt;
    const float var = rb[0] / cnt - mu * mu;
    statsKQ[(which * nB + b) * 2 + 0] = mu;
    statsKQ[(which * nB + b) * 2 + 1] = rsqrtf(var + 1e-5f);
  }
}

// ---------------- Kernel 3: LN stats for V (per b,h) -----------------------
__global__ __launch_bounds__(256) void stats_v_kernel(const float* __restrict__ P,
                                                      float* __restrict__ statsV) {
  const int b = blockIdx.x, h = blockIdx.y, which = blockIdx.z;  // which: 0=V1, 1=V2
  const float* PV = P + (size_t)(4 + which) * SLAB + (size_t)b * nN * nHD + h * nD;
  float s = 0.f, ss = 0.f;
  for (int i = threadIdx.x; i < nN * nD; i += 256) {
    const int n = i >> 6, d = i & 63;
    const float v = PV[(size_t)n * nHD + d];
    s += v; ss += v * v;
  }
  __shared__ float ra[256], rb[256];
  ra[threadIdx.x] = s; rb[threadIdx.x] = ss;
  __syncthreads();
  for (int st = 128; st > 0; st >>= 1) {
    if (threadIdx.x < st) { ra[threadIdx.x] += ra[threadIdx.x + st]; rb[threadIdx.x] += rb[threadIdx.x + st]; }
    __syncthreads();
  }
  if (threadIdx.x == 0) {
    const float cnt = (float)(nN * nD);
    const float mu = ra[0] / cnt;
    const float var = rb[0] / cnt - mu * mu;
    statsV[((which * nB + b) * nH + h) * 2 + 0] = mu;
    statsV[((which * nB + b) * nH + h) * 2 + 1] = rsqrtf(var + 1e-5f);
  }
}

// ---------------- Kernel 4: apply V layernorm, emit f16 [B,H,N,D] ----------
__global__ __launch_bounds__(256) void vnorm_kernel(const float* __restrict__ P,
    const float* __restrict__ statsV, const float* __restrict__ vnw,
    const float* __restrict__ vnb, f16* __restrict__ V1h, f16* __restrict__ V2h) {
  const int idx = blockIdx.x * 256 + threadIdx.x;   // 0 .. 2*262144-1
  constexpr int per = nB * nH * nN * nD;            // 262144
  const int which = idx / per;
  const int r = idx % per;
  const int b = r >> 15, h = (r >> 13) & 3, n = (r >> 6) & 127, d = r & 63;
  const float* PV = P + (size_t)(4 + which) * SLAB;
  const float v = PV[((size_t)b * nN + n) * nHD + h * nD + d];
  const float mu  = statsV[((which * nB + b) * nH + h) * 2 + 0];
  const float inv = statsV[((which * nB + b) * nH + h) * 2 + 1];
  const float o = (v - mu) * inv * vnw[n * nD + d] + vnb[n * nD + d];
  f16* Vh = which ? V2h : V1h;
  Vh[((size_t)(b * nH + h) * nN + n) * nD + d] = (f16)o;
}

// ---------------- Kernel 5: fused per-(b,f,h) attention block --------------
// Dynamic LDS layout (221456 B, < 320 KB/WGP):
//   [0      , 102400): S2 f32[160][160]   (phase 2 output)
//       overlay during phases 0-1: Qh/Kh/AQh/AKh f16[160][64] (81920 B)
//   [102400 , 153600): S  f16[160][160]   (phase 1 output)
//   [153600 , 154880): colMax/colSum f32[160] each
//   [154880 , 221456): AAW f32[129*129]   (async-DMA staged aa_w)
constexpr int OFF_S   = nMP * nMP * 4;          // 102400
constexpr int OFF_CM  = OFF_S + nMP * nMP * 2;  // 153600
constexpr int OFF_AAW = OFF_CM + 2 * nMP * 4;   // 154880
constexpr int ATTN_SMEM = OFF_AAW + 16644 * 4;  // 221456

__global__ __launch_bounds__(256) void attn_kernel(
    const float* __restrict__ P, const float* __restrict__ statsKQ,
    const float* __restrict__ knw, const float* __restrict__ knb,
    const float* __restrict__ qnw, const float* __restrict__ qnb,
    const float* __restrict__ akw, const float* __restrict__ akb,
    const float* __restrict__ aqw, const float* __restrict__ aqb,
    const float* __restrict__ aaw, const float* __restrict__ aab,
    float* __restrict__ outA1, float* __restrict__ outA2,
    f16* __restrict__ A1h, f16* __restrict__ A2h) {
  extern __shared__ char smem[];
  float* S2  = (float*)smem;                 // [160][160] f32
  f16*   S   = (f16*)(smem + OFF_S);         // [160][160] f16
  f16*   Qh  = (f16*)smem;                   // [160][64] overlays S2
  f16*   Kh  = Qh + nMP * nD;
  f16*   AQh = Kh + nMP * nD;                // aq_w staged f16, zero-padded rows
  f16*   AKh = AQh + nMP * nD;
  float* colMax = (float*)(smem + OFF_CM);
  float* colSum = colMax + nMP;
  float* AAW = (float*)(smem + OFF_AAW);     // aa_w staged fp32 [129][129]

  const int f = blockIdx.x, h = blockIdx.y, b = blockIdx.z;
  const int tid = threadIdx.x, lane = tid & 31, wv = tid >> 5;

  // Kick off async DMA of aa_w into LDS; overlaps with phases 0-1.
  {
    constexpr int nvec = (nM * nM) / 4;      // 4160 16-byte vectors
    for (int i4 = tid; i4 < nvec; i4 += 256) {
      const uint32_t la = (uint32_t)(uintptr_t)(AAW + i4 * 4);
      async_g2lds_b128(la, aaw + i4 * 4);
    }
    if (tid == 0) {                          // tail element 16640
      const uint32_t la = (uint32_t)(uintptr_t)(AAW + nM * nM - 1);
      async_g2lds_b32(la, aaw + nM * nM - 1);
    }
  }

  const float* PK1 = P;
  const float* PK2 = P + (size_t)1 * SLAB;
  const float* PQ1 = P + (size_t)2 * SLAB;
  const float* PQ2 = P + (size_t)3 * SLAB;
  const float muK = statsKQ[(0 * nB + b) * 2 + 0], invK = statsKQ[(0 * nB + b) * 2 + 1];
  const float muQ = statsKQ[(1 * nB + b) * 2 + 0], invQ = statsKQ[(1 * nB + b) * 2 + 1];

  // Phase 0: normalized Q/K rows (row 0 = m1 proj, rows 1..128 = m2 proj) and
  // zero-padded f16 copies of aq_w/ak_w.
  for (int idx = tid; idx < nMP * nD; idx += 256) {
    const int n = idx >> 6, d = idx & 63;
    float qv = 0.f, kv = 0.f;
    if (n < nM) {
      const int e = h * nD + d;
      const float qraw = (n == 0) ? PQ1[((size_t)b * nN + f) * nHD + e]
                                  : PQ2[((size_t)b * nN + (n - 1)) * nHD + e];
      const float kraw = (n == 0) ? PK1[((size_t)b * nN + f) * nHD + e]
                                  : PK2[((size_t)b * nN + (n - 1)) * nHD + e];
      const size_t wi = (((size_t)f * nH + h) * nM + n) * nD + d;
      qv = (qraw - muQ) * invQ * qnw[wi] + qnb[wi];
      kv = (kraw - muK) * invK * knw[wi] + knb[wi];
    }
    Qh[idx] = (f16)qv;
    Kh[idx] = (f16)kv;
    AQh[idx] = (n < nM) ? (f16)aqw[idx] : (f16)0.0f;  // idx == n*64+d
    AKh[idx] = (n < nM) ? (f16)akw[idx] : (f16)0.0f;
  }
  __syncthreads();

  // Phase 1: S = elu(Q aq^T + K ak^T + aq_b + ak_b), 10x10 tiles of 16x16, K=64
  for (int t = wv; t < 100; t += 8) {
    const int i = t / 10, j = t % 10;
    const int arow = i * 16 + (lane & 15);
    const int mcol = j * 16 + (lane & 15);
    const f16* qrow = Qh + arow * nD;
    const f16* krow = Kh + arow * nD;
    const f16* aqc = AQh + mcol * nD;        // rows >= 129 are zero
    const f16* akc = AKh + mcol * nD;
    v8f acc = {};
#pragma unroll
    for (int kk = 0; kk < nD; kk += 32) {
      acc = wmma_f16(a_frag_f16(qrow, kk, lane), b_frag_f16(aqc, kk, lane), acc);
      acc = wmma_f16(a_frag_f16(krow, kk, lane), b_frag_f16(akc, kk, lane), acc);
    }
    const int rbase = i * 16 + ((lane & 16) ? 8 : 0);
#pragma unroll
    for (int r = 0; r < 8; ++r) {
      const int n = rbase + r;
      float v = 0.f;
      if (mcol < nM && n < nM) {
        v = acc[r] + aqb[mcol] + akb[mcol];
        v = (v > 0.f) ? v : (expf(v) - 1.f);   // elu, alpha=1
      }
      S[n * nMP + mcol] = (f16)v;              // pad rows/cols -> 0
    }
  }
  wait_async0();      // aa_w DMA must have landed in LDS
  __syncthreads();

  // Phase 2: S2 = S aa^T + aa_b, 10x9 tiles, K=160 (cols >=129 of S are zero)
  for (int t = wv; t < 90; t += 8) {
    const int i = t / 9, j = t % 9;
    const int arow = i * 16 + (lane & 15);
    const int og = j * 16 + (lane & 15);
    const f16* srow = S + arow * nMP;
    v8f acc = {};
#pragma unroll
    for (int kk = 0; kk < nMP; kk += 32)
      acc = wmma_f16(a_frag_f16(srow, kk, lane), b_frag_aa_lds(AAW, og, kk, lane), acc);
    const int rbase = i * 16 + ((lane & 16) ? 8 : 0);
    const float bo = (og < nM) ? aab[og] : 0.f;
#pragma unroll
    for (int r = 0; r < 8; ++r)
      S2[(rbase + r) * nMP + og] = acc[r] + bo;
  }
  __syncthreads();

  // Phase 3: softmax stats per column o (over n = 0..128 only)
  if (tid < nM) {
    float mx = -3.4e38f;
    for (int n = 0; n < nM; ++n) mx = fmaxf(mx, S2[n * nMP + tid]);
    float sm = 0.f;
    for (int n = 0; n < nM; ++n) sm += expf(S2[n * nMP + tid] - mx);
    colMax[tid] = mx; colSum[tid] = sm;
  }
  __syncthreads();

  // Phase 4: A_1[b,h,f,m] = softmax_n(S2)[n=128, m] ; A_2[b,h,f,n] = softmax_n(S2)[n, o=128]
  if (tid < nN) {
    const size_t base = (((size_t)b * nH + h) * nN + f) * nN;
    const float a1 = expf(S2[nN * nMP + tid] - colMax[tid]) / colSum[tid];
    outA1[base + tid] = a1;  A1h[base + tid] = (f16)a1;
    const float a2 = expf(S2[tid * nMP + nN] - colMax[nN]) / colSum[nN];
    outA2[base + tid] = a2;  A2h[base + tid] = (f16)a2;
  }
}

// ---------------- Kernel 6: E1 = A2 x V1, E2 = A1 x V2 (batched, K=128) ----
__global__ __launch_bounds__(32) void av_gemm_kernel(
    const f16* __restrict__ A1h, const f16* __restrict__ A2h,
    const f16* __restrict__ V1h, const f16* __restrict__ V2h,
    f16* __restrict__ E1h, f16* __restrict__ E2h) {
  const int lane = threadIdx.x & 31;
  const int rt = blockIdx.x;        // 0..7  f-tiles
  const int ct = blockIdx.y;        // 0..3  d-tiles
  const int z = blockIdx.z;         // 0..63
  const int which = z >> 5;         // 0: E1=A2*V1, 1: E2=A1*V2
  const int bh = z & 31;
  const f16* A = which ? A1h : A2h;
  const f16* V = which ? V2h : V1h;
  f16* E = which ? E2h : E1h;
  const int b = bh >> 2, h = bh & 3;

  const int frow = rt * 16 + (lane & 15);
  const int dcol = ct * 16 + (lane & 15);
  const f16* arow = A + ((size_t)bh * nN + frow) * nN;
  const f16* vbase = V + (size_t)bh * nN * nD;

  v8f acc = {};
#pragma unroll
  for (int kk = 0; kk < nN; kk += 32)
    acc = wmma_f16(a_frag_f16(arow, kk, lane), b_frag_f16_strided(vbase, nD, dcol, kk, lane), acc);

  const int rbase = rt * 16 + ((lane & 16) ? 8 : 0);
#pragma unroll
  for (int r = 0; r < 8; ++r)
    E[((size_t)b * nN + rbase + r) * nHD + h * nD + dcol] = (f16)acc[r];  // [b, f, h*64+d]
}

// ---------------- Kernel 7: out = relu(E l1_w^T + l1_b) --------------------
__global__ __launch_bounds__(256) void final_gemm_kernel(
    const f16* __restrict__ E1h, const f16* __restrict__ E2h,
    const float* __restrict__ l1w, const float* __restrict__ l1b,
    float* __restrict__ out) {
  const int lane = threadIdx.x & 31, wv = threadIdx.x >> 5;
  const int rt = blockIdx.x;              // 0..63
  const int ct = blockIdx.y * 8 + wv;     // 0..15
  const int which = blockIdx.z;
  const f16* E = which ? E2h : E1h;
  float* O = out + (size_t)which * (nB * nN * nIN);

  const int row = rt * 16 + (lane & 15);
  const int col = ct * 16 + (lane & 15);
  const f16* rowp = E + (size_t)row * nHD;
  const float* colp = l1w + (size_t)col * nIN;

  v8f acc = {};
#pragma unroll
  for (int kk = 0; kk < nHD; kk += 32)
    acc = wmma_f16(a_frag_f16(rowp, kk, lane), b_frag_f32(colp, kk, lane), acc);

  const int rbase = rt * 16 + ((lane & 16) ? 8 : 0);
#pragma unroll
  for (int r = 0; r < 8; ++r) {
    const float v = acc[r] + l1b[col];
    O[(size_t)(rbase + r) * nIN + col] = fmaxf(v, 0.f);
  }
}

// ---------------- Host launcher --------------------------------------------
extern "C" void kernel_launch(void* const* d_in, const int* in_sizes, int n_in,
                              void* d_out, int out_size, void* d_ws, size_t ws_size,
                              hipStream_t stream) {
  (void)in_sizes; (void)n_in; (void)out_size; (void)ws_size;
  const float* m1  = (const float*)d_in[0];
  const float* m2  = (const float*)d_in[1];
  const float* kw  = (const float*)d_in[2];
  const float* kb  = (const float*)d_in[3];
  const float* qw  = (const float*)d_in[4];
  const float* qb  = (const float*)d_in[5];
  const float* vw  = (const float*)d_in[6];
  const float* vb  = (const float*)d_in[7];
  const float* knw = (const float*)d_in[8];
  const float* knb = (const float*)d_in[9];
  const float* qnw = (const float*)d_in[10];
  const float* qnb = (const float*)d_in[11];
  const float* vnw = (const float*)d_in[12];
  const float* vnb = (const float*)d_in[13];
  const float* akw = (const float*)d_in[14];
  const float* akb = (const float*)d_in[15];
  const float* aqw = (const float*)d_in[16];
  const float* aqb = (const float*)d_in[17];
  const float* aaw = (const float*)d_in[18];
  const float* aab = (const float*)d_in[19];
  const float* l1w = (const float*)d_in[20];
  const float* l1b = (const float*)d_in[21];

  // Workspace carve-up (~10.5 MB total)
  char* w = (char*)d_ws;
  float* P        = (float*)(w);                // 6 slabs x 1 MB (PK1,PK2,PQ1,PQ2,PV1,PV2)
  float* statsKQ  = (float*)(w + 6291456);      // 32 floats  {mu,inv} per (K/Q, b)
  float* statsV   = (float*)(w + 6291584);      // 128 floats {mu,inv} per (V1/V2, b, h)
  f16*   V1h      = (f16*)(w + 6292096);        // [B,H,N,D] f16
  f16*   V2h      = (f16*)(w + 6816384);
  f16*   A1h      = (f16*)(w + 7340672);        // [B,H,N,N] f16
  f16*   A2h      = (f16*)(w + 8389248);
  f16*   E1h      = (f16*)(w + 9437824);        // [B*N, 256] f16
  f16*   E2h      = (f16*)(w + 9962112);

  // Output: E_1 | E_2 | A_1 | A_2 (flat, fp32)
  float* outE  = (float*)d_out;
  float* outA1 = outE + 2 * nB * nN * nIN;      // +524288
  float* outA2 = outA1 + nB * nH * nN * nN;     // +524288

  proj_gemm_kernel<<<dim3(64, 2, 6), 256, 0, stream>>>(m1, m2, kw, kb, qw, qb, vw, vb, P);
  stats_kq_kernel<<<dim3(8, 2, 1), 256, 0, stream>>>(P, statsKQ);
  stats_v_kernel<<<dim3(8, 4, 2), 256, 0, stream>>>(P, statsV);
  vnorm_kernel<<<dim3(2048, 1, 1), 256, 0, stream>>>(P, statsV, vnw, vnb, V1h, V2h);
  attn_kernel<<<dim3(128, 4, 8), 256, ATTN_SMEM, stream>>>(
      P, statsKQ, knw, knb, qnw, qnb, akw, akb, aqw, aqb, aaw, aab,
      outA1, outA2, A1h, A2h);
  av_gemm_kernel<<<dim3(8, 4, 64), 32, 0, stream>>>(A1h, A2h, V1h, V2h, E1h, E2h);
  final_gemm_kernel<<<dim3(64, 2, 2), 256, 0, stream>>>(E1h, E2h, l1w, l1b, outE);
}
